// WindowAttention_81913616269963
// MI455X (gfx1250) — matrix operand order repaired and checked by reading, not compiled, not run on hardware
//
#include <hip/hip_runtime.h>
#include <hip/hip_bf16.h>

typedef __bf16 bf16_t;
typedef __attribute__((ext_vector_type(16))) __bf16 v16bf;
typedef __attribute__((ext_vector_type(8)))  float  v8f;

#define WIN_N   49
#define NPAD    64
#define DIM     256
#define HEADS   8
#define HD      32
#define SCALE_F 0.17677669529663687f   // 32^-0.5

#define XS_STRIDE 264    // halves; 528B rows, 16B aligned
#define VT_STRIDE 72     // halves; 144B rows, 16B aligned
#define PS_STRIDE 72

// LDS carve: xs | qs | ks | vT | ps[8 waves]
#define OFF_XS 0
#define OFF_QS (NPAD * XS_STRIDE)
#define OFF_KS (2 * NPAD * XS_STRIDE)
#define OFF_VT (3 * NPAD * XS_STRIDE)
#define OFF_PS (OFF_VT + DIM * VT_STRIDE)
#define SMEM_HALVES (OFF_PS + HEADS * NPAD * PS_STRIDE)
#define SMEM_FUSED (SMEM_HALVES * 2)   // 211,968 bytes

union FragU { v16bf v; uint4 u[2]; };

// A-matrix 16x32 bf16 fragment: lane m = lane&15, g = lane>>4.
// elements 0..7 = K[8g .. 8g+7], elements 8..15 = K[16+8g .. 16+8g+7]
__device__ __forceinline__ v16bf load_frag_A(const bf16_t* rowPtr, int g) {
  FragU f;
  f.u[0] = *(const uint4*)(rowPtr + 8 * g);
  f.u[1] = *(const uint4*)(rowPtr + 16 + 8 * g);
  return f.v;
}
// B-matrix 32x16 bf16 fragment: lane n = lane&15, g = lane>>4.
// elements 0..7 = K[8g .. 8g+7], elements 8..15 = K[16+8g .. 16+8g+7]
__device__ __forceinline__ v16bf load_frag_B(const bf16_t* rowPtr, int g) {
  FragU f;
  f.u[0] = *(const uint4*)(rowPtr + 8 * g);
  f.u[1] = *(const uint4*)(rowPtr + 16 + 8 * g);
  return f.v;
}

__device__ __forceinline__ v8f wmma_bf16(v16bf a, v16bf b, v8f c) {
  return __builtin_amdgcn_wmma_f32_16x16x32_bf16(false, a, false, b, (short)0, c,
                                                 false, false);
}

// ---------------------------------------------------------------------------
// Kernel 0: convert the four 256x256 fp32 weights to bf16 (stay hot in L2)
// ---------------------------------------------------------------------------
__global__ __launch_bounds__(256)
void k_cvt_w(const float* __restrict__ wq, const float* __restrict__ wk,
             const float* __restrict__ wv, const float* __restrict__ wp,
             bf16_t* __restrict__ oq, bf16_t* __restrict__ ok,
             bf16_t* __restrict__ ov, bf16_t* __restrict__ op) {
  int i = blockIdx.x * blockDim.x + threadIdx.x;
  if (i < DIM * DIM) {
    oq[i] = (bf16_t)wq[i];
    ok[i] = (bf16_t)wk[i];
    ov[i] = (bf16_t)wv[i];
    op[i] = (bf16_t)wp[i];
  }
}

// ---------------------------------------------------------------------------
// Kernel 1: fused per-window  QKV projection + attention  (1 block = 1 window,
//           8 waves = 8 heads).  ctx (bf16) is the only HBM intermediate.
// ---------------------------------------------------------------------------
__global__ __launch_bounds__(256, 1)
void k_fused(const float* __restrict__ x,
             const float* __restrict__ bias_table,
             const int*   __restrict__ rel_idx,
             const bf16_t* __restrict__ wq, const bf16_t* __restrict__ wk,
             const bf16_t* __restrict__ wv,
             const float* __restrict__ bq, const float* __restrict__ bk,
             const float* __restrict__ bv,
             bf16_t* __restrict__ ctx) {
  extern __shared__ char smemRaw[];
  bf16_t* sm  = (bf16_t*)smemRaw;
  bf16_t* xs  = sm + OFF_XS;   // [64][264] bf16 x-window (zero-padded rows)
  bf16_t* qs  = sm + OFF_QS;   // [64][264] q * SCALE
  bf16_t* ks  = sm + OFF_KS;   // [64][264] k
  bf16_t* vT  = sm + OFF_VT;   // [256][72] v transposed (dim-major)
  bf16_t* ps  = sm + OFF_PS;   // [8][64][72] per-wave softmax P

  const int b    = blockIdx.x;
  const int tid  = threadIdx.x;
  const int w    = tid >> 5;     // wave id == head id in stage 3
  const int lane = tid & 31;
  const int n16  = lane & 15;
  const int g    = lane >> 4;

  // ---- Stage 1: x (fp32) -> xs (bf16), rows 49..63 zeroed ---------------
  const float* xw = x + (size_t)b * WIN_N * DIM;
  #pragma unroll 1
  for (int j = 0; j < NPAD; ++j) {
    float v = (j < WIN_N) ? xw[j * DIM + tid] : 0.0f;
    xs[j * XS_STRIDE + tid] = (bf16_t)v;
  }
  __syncthreads();

  // ---- Stage 2: q/k/v = xs @ W^T + b  (48 col-blocks, 6 per wave) -------
  #pragma unroll 1
  for (int r = 0; r < 6; ++r) {
    const int cb  = w + 8 * r;        // 0..47
    const int sel = cb >> 4;          // 0:q 1:k 2:v
    const int ob  = (cb & 15) * 16;   // output column base
    const bf16_t* W  = (sel == 0) ? wq : (sel == 1) ? wk : wv;
    const float*  bi = (sel == 0) ? bq : (sel == 1) ? bk : bv;

    v8f z = (v8f)(0.0f);
    v8f acc[4] = {z, z, z, z};
    #pragma unroll 2
    for (int kk = 0; kk < DIM; kk += 32) {
      __builtin_prefetch(W + (ob + n16) * DIM + kk + 64, 0, 1);
      v16bf bf = load_frag_B(W + (ob + n16) * DIM + kk, g);
      #pragma unroll
      for (int rb = 0; rb < 4; ++rb) {
        v16bf af = load_frag_A(xs + (rb * 16 + n16) * XS_STRIDE + kk, g);
        acc[rb] = wmma_bf16(af, bf, acc[rb]);
      }
    }
    const int   col  = ob + n16;
    const float bcol = bi[col];
    #pragma unroll
    for (int rb = 0; rb < 4; ++rb) {
      #pragma unroll
      for (int i = 0; i < 8; ++i) {
        int   tok = rb * 16 + i + 8 * g;   // C/D layout: M = i + 8g
        float vv  = acc[rb][i] + bcol;
        if (sel == 0)      qs[tok * XS_STRIDE + col] = (bf16_t)(vv * SCALE_F);
        else if (sel == 1) ks[tok * XS_STRIDE + col] = (bf16_t)vv;
        else               vT[col * VT_STRIDE + tok] = (bf16_t)vv;
      }
    }
  }
  __syncthreads();

  // ---- Stage rel_idx / bias_table into the (now dead) xs region ---------
  int*   relS = (int*)xs;                         // 2401 ints  (9604 B)
  float* btS  = (float*)(xs + 4808);              // 1352 f32   (5408 B)
  #pragma unroll 1
  for (int i = tid; i < WIN_N * WIN_N; i += 256) relS[i] = rel_idx[i];
  #pragma unroll 1
  for (int i = tid; i < 169 * HEADS; i += 256)   btS[i] = bias_table[i];
  __syncthreads();

  // ---- Stage 3: attention, wave w == head h -----------------------------
  const int h  = w;
  bf16_t*  pw = ps + w * NPAD * PS_STRIDE;

  // q A-fragments (K = head dim = 32, single WMMA per tile)
  v16bf aq[4];
  #pragma unroll
  for (int rb = 0; rb < 4; ++rb)
    aq[rb] = load_frag_A(qs + (rb * 16 + n16) * XS_STRIDE + h * HD, g);

  v8f z = (v8f)(0.0f);
  v8f s[4][4];
  #pragma unroll
  for (int cb = 0; cb < 4; ++cb) {
    v16bf bkf = load_frag_B(ks + (cb * 16 + n16) * XS_STRIDE + h * HD, g);
    #pragma unroll
    for (int rb = 0; rb < 4; ++rb) s[rb][cb] = wmma_bf16(aq[rb], bkf, z);
  }

  // bias + mask + softmax (rows live in one 16-lane half -> width-16 shfl)
  #pragma unroll
  for (int rb = 0; rb < 4; ++rb) {
    #pragma unroll
    for (int iv = 0; iv < 8; ++iv) {
      const int i = rb * 16 + iv + 8 * g;   // token row
      float m = -3.0e38f;
      #pragma unroll
      for (int cb = 0; cb < 4; ++cb) {
        const int j = cb * 16 + n16;
        float sv = s[rb][cb][iv];
        if (i < WIN_N && j < WIN_N) sv += btS[relS[i * WIN_N + j] * HEADS + h];
        else                        sv  = -1.0e30f;
        s[rb][cb][iv] = sv;
        m = fmaxf(m, sv);
      }
      #pragma unroll
      for (int d = 1; d < 16; d <<= 1) m = fmaxf(m, __shfl_xor(m, d, 16));
      float sum = 0.0f;
      #pragma unroll
      for (int cb = 0; cb < 4; ++cb) {
        float e = __expf(s[rb][cb][iv] - m);
        s[rb][cb][iv] = e;
        sum += e;
      }
      #pragma unroll
      for (int d = 1; d < 16; d <<= 1) sum += __shfl_xor(sum, d, 16);
      const float rcp = 1.0f / sum;
      #pragma unroll
      for (int cb = 0; cb < 4; ++cb)
        pw[i * PS_STRIDE + cb * 16 + n16] = (bf16_t)(s[rb][cb][iv] * rcp);
    }
  }
  // intra-wave LDS RAW: compiler inserts s_wait_dscnt; no block barrier needed

  // O = P @ V  (K = 64 padded tokens; padded P columns are exactly 0)
  v16bf ap[4][2];
  #pragma unroll
  for (int rb = 0; rb < 4; ++rb)
    #pragma unroll
    for (int kq = 0; kq < 2; ++kq)
      ap[rb][kq] = load_frag_A(pw + (rb * 16 + n16) * PS_STRIDE + kq * 32, g);

  v8f o[4][2] = {{z, z}, {z, z}, {z, z}, {z, z}};
  #pragma unroll
  for (int cb = 0; cb < 2; ++cb) {
    #pragma unroll
    for (int kq = 0; kq < 2; ++kq) {
      v16bf bvf =
          load_frag_B(vT + (h * HD + cb * 16 + n16) * VT_STRIDE + kq * 32, g);
      #pragma unroll
      for (int rb = 0; rb < 4; ++rb) o[rb][cb] = wmma_bf16(ap[rb][kq], bvf, o[rb][cb]);
    }
  }

  // write context (bf16), head h owns columns [h*32, h*32+32)
  bf16_t* cw = ctx + (size_t)b * NPAD * DIM;
  #pragma unroll
  for (int rb = 0; rb < 4; ++rb) {
    #pragma unroll
    for (int cb = 0; cb < 2; ++cb) {
      const int col = h * HD + cb * 16 + n16;
      #pragma unroll
      for (int iv = 0; iv < 8; ++iv) {
        const int tok = rb * 16 + iv + 8 * g;
        cw[tok * DIM + col] = (bf16_t)o[rb][cb][iv];
      }
    }
  }
}

// ---------------------------------------------------------------------------
// Kernel 2: out = ctx @ Wp^T + bp  (fp32 output, tokens < 49 only)
// ---------------------------------------------------------------------------
__global__ __launch_bounds__(256)
void k_proj(const bf16_t* __restrict__ ctx, const bf16_t* __restrict__ wp,
            const float* __restrict__ bp, float* __restrict__ out) {
  const int b    = blockIdx.x;
  const int tid  = threadIdx.x;
  const int w    = tid >> 5;
  const int lane = tid & 31;
  const int n16  = lane & 15;
  const int g    = lane >> 4;

  const bf16_t* cw = ctx + (size_t)b * NPAD * DIM;

  #pragma unroll 1
  for (int r = 0; r < 2; ++r) {
    const int ob = (w * 2 + r) * 16;
    v8f z = (v8f)(0.0f);
    v8f acc[4] = {z, z, z, z};
    #pragma unroll 2
    for (int kk = 0; kk < DIM; kk += 32) {
      v16bf bf = load_frag_B(wp + (ob + n16) * DIM + kk, g);
      #pragma unroll
      for (int rb = 0; rb < 4; ++rb) {
        v16bf af = load_frag_A(cw + (rb * 16 + n16) * DIM + kk, g);
        acc[rb] = wmma_bf16(af, bf, acc[rb]);
      }
    }
    const int   col  = ob + n16;
    const float bcol = bp[col];
    #pragma unroll
    for (int rb = 0; rb < 4; ++rb) {
      #pragma unroll
      for (int iv = 0; iv < 8; ++iv) {
        const int tok = rb * 16 + iv + 8 * g;
        if (tok < WIN_N)
          out[((size_t)b * WIN_N + tok) * DIM + col] = acc[rb][iv] + bcol;
      }
    }
  }
}

// ---------------------------------------------------------------------------
extern "C" void kernel_launch(void* const* d_in, const int* in_sizes, int n_in,
                              void* d_out, int out_size, void* d_ws, size_t ws_size,
                              hipStream_t stream) {
  const float* x          = (const float*)d_in[0];
  const float* bias_table = (const float*)d_in[1];
  const float* Wq         = (const float*)d_in[2];
  const float* bq         = (const float*)d_in[3];
  const float* Wk         = (const float*)d_in[4];
  const float* bk         = (const float*)d_in[5];
  const float* Wv         = (const float*)d_in[6];
  const float* bv         = (const float*)d_in[7];
  const float* Wp         = (const float*)d_in[8];
  const float* bp         = (const float*)d_in[9];
  const int*   rel_idx    = (const int*)d_in[10];
  float*       out        = (float*)d_out;

  char* ws = (char*)d_ws;
  bf16_t* wq_h = (bf16_t*)ws;             ws += DIM * DIM * sizeof(bf16_t);
  bf16_t* wk_h = (bf16_t*)ws;             ws += DIM * DIM * sizeof(bf16_t);
  bf16_t* wv_h = (bf16_t*)ws;             ws += DIM * DIM * sizeof(bf16_t);
  bf16_t* wp_h = (bf16_t*)ws;             ws += DIM * DIM * sizeof(bf16_t);
  bf16_t* ctx  = (bf16_t*)ws;             // 4096 * 64 * 256 bf16 = 128 MiB

  hipFuncSetAttribute((const void*)k_fused,
                      hipFuncAttributeMaxDynamicSharedMemorySize, SMEM_FUSED);

  k_cvt_w<<<DIM * DIM / 256, 256, 0, stream>>>(Wq, Wk, Wv, Wp,
                                               wq_h, wk_h, wv_h, wp_h);
  k_fused<<<4096, 256, SMEM_FUSED, stream>>>(x, bias_table, rel_idx,
                                             wq_h, wk_h, wv_h, bq, bk, bv, ctx);
  k_proj<<<4096, 256, 0, stream>>>(ctx, wp_h, bp, out);
}
